// LayerGCN_71416716198486
// MI455X (gfx1250) — matrix operand order repaired and checked by reading, not compiled, not run on hardware
//
#include <hip/hip_runtime.h>
#include <hip/hip_bf16.h>

#define NU  200000
#define NI  100000
#define NN  (NU + NI)
#define DIM 64
#define NE  2000000

typedef __attribute__((ext_vector_type(16))) _Float16 v16h;
typedef __attribute__((ext_vector_type(8)))  float    v8f;

// ---------------------------------------------------------------- utilities
__global__ void zero_kernel(float* __restrict__ p, long n) {
  long i = (long)blockIdx.x * blockDim.x + threadIdx.x;
  if (i < n) p[i] = 0.0f;
}

// deg_u/deg_i via float atomics (segment_sum of ones)
__global__ void degree_kernel(const int* __restrict__ rows, const int* __restrict__ cols,
                              float* __restrict__ du, float* __restrict__ di) {
  int e = blockIdx.x * blockDim.x + threadIdx.x;
  if (e < NE) {
    atomicAdd(&du[rows[e]], 1.0f);
    atomicAdd(&di[cols[e]], 1.0f);
  }
}

// vals[e] = (1e-7+deg_u)^-0.5 * (1e-7+deg_i)^-0.5
__global__ void vals_kernel(const int* __restrict__ rows, const int* __restrict__ cols,
                            const float* __restrict__ du, const float* __restrict__ di,
                            float* __restrict__ vals) {
  int e = blockIdx.x * blockDim.x + threadIdx.x;
  if (e < NE)
    vals[e] = rsqrtf(du[rows[e]] + 1e-7f) * rsqrtf(di[cols[e]] + 1e-7f);
}

// one wave (32 lanes) per node: ||ego_node||, clamped at eps
__global__ void ego_norm_kernel(const float* __restrict__ ue, const float* __restrict__ ie,
                                float* __restrict__ enorm) {
  int wid  = (blockIdx.x * blockDim.x + threadIdx.x) >> 5;
  int lane = threadIdx.x & 31;
  if (wid >= NN) return;
  const float* p = (wid < NU) ? (ue + (size_t)wid * DIM)
                              : (ie + (size_t)(wid - NU) * DIM);
  float2 f = *(const float2*)(p + lane * 2);
  float s = f.x * f.x + f.y * f.y;
#pragma unroll
  for (int off = 16; off > 0; off >>= 1) s += __shfl_xor(s, off, 32);
  if (lane == 0) enorm[wid] = fmaxf(sqrtf(s), 1e-8f);
}

// SpMM: one wave per directed edge. Gather src row (coalesced 256B), scale,
// atomic-scatter into dst row. x rows are L2-resident (77MB < 192MB L2).
__global__ void spmm_kernel(const int* __restrict__ rows, const int* __restrict__ cols,
                            const float* __restrict__ vals,
                            const float* __restrict__ xu, const float* __restrict__ xi,
                            float* __restrict__ xout) {
  long tid = (long)blockIdx.x * blockDim.x + threadIdx.x;
  long de  = tid >> 5;
  int lane = threadIdx.x & 31;
  if (de >= 2L * NE) return;
  int e   = (int)(de >> 1);
  int dir = (int)(de & 1);

  if (lane == 0 && e + 4096 < NE) {            // stream edge lists ahead of use
    __builtin_prefetch(&rows[e + 4096], 0, 1); // -> global_prefetch_b8
    __builtin_prefetch(&cols[e + 4096], 0, 1);
    __builtin_prefetch(&vals[e + 4096], 0, 1);
  }

  int   r = rows[e], c = cols[e];
  float v = vals[e];
  int src = dir ? r : (c + NU);
  int dst = dir ? (c + NU) : r;
  const float* xs = (src < NU) ? (xu + (size_t)src * DIM)
                               : (xi + (size_t)(src - NU) * DIM);
  float2 f = *(const float2*)(xs + lane * 2);
  float* o = xout + (size_t)dst * DIM + lane * 2;
  atomicAdd(o,     v * f.x);
  atomicAdd(o + 1, v * f.y);
}

// Cosine reweighting: one wave per 16-node tile.
//   diag(X * Ego^T) = per-node dot(x, ego)
//   diag(X * X^T)   = per-node ||x||^2
// computed via v_wmma_f32_16x16x32_f16 (K=64 in two chunks), diagonal
// extracted through LDS per the 16x16 f32 C/D lane layout.
__global__ void __launch_bounds__(32)
reweight_kernel(float* __restrict__ x, const float* __restrict__ ue,
                const float* __restrict__ ie, const float* __restrict__ enorm,
                float* __restrict__ acc) {
  __shared__ float sdot[16];
  __shared__ float snsq[16];
  __shared__ float sw[16];

  int n0   = blockIdx.x * 16;
  int lane = threadIdx.x;           // blockDim.x == 32, EXEC all-1s for WMMA
  int m    = lane & 15;             // A row index == B col index for this lane
  int kb   = (lane >> 4) * 8;       // A-layout K base for upper half-wave

  int nodeB = n0 + m;
  const float* xrowA = x + (size_t)nodeB * DIM;          // A rows = x rows
  const float* erowB = (nodeB < NU) ? (ue + (size_t)nodeB * DIM)
                                    : (ie + (size_t)(nodeB - NU) * DIM);
  const float* xrowB = x + (size_t)nodeB * DIM;          // B cols = x rows (for X*X^T)

  v8f cdot = {0.f,0.f,0.f,0.f,0.f,0.f,0.f,0.f};
  v8f cnsq = {0.f,0.f,0.f,0.f,0.f,0.f,0.f,0.f};

#pragma unroll
  for (int c = 0; c < 2; ++c) {                 // K = 64 in two K=32 chunks
    v16h a{}, bE{}, bX{};
#pragma unroll
    for (int j = 0; j < 16; ++j) {
      // 16-bit A 16x32 layout: lane<16 -> K {0..7,16..23}; lane>=16 -> {8..15,24..31}
      int kA = 32 * c + kb + (j & 7) + ((j >> 3) << 4);
      // B uses same K order as A: lane<16 -> K {0..7,16..23}; lane>=16 -> {8..15,24..31}
      int kB = 32 * c + kb + (j & 7) + ((j >> 3) << 4);
      a[j]  = (_Float16)xrowA[kA];
      bE[j] = (_Float16)erowB[kB];
      bX[j] = (_Float16)xrowB[kB];
    }
    cdot = __builtin_amdgcn_wmma_f32_16x16x32_f16(false, a, false, bE,
                                                  (short)0, cdot, false, false);
    cnsq = __builtin_amdgcn_wmma_f32_16x16x32_f16(false, a, false, bX,
                                                  (short)0, cnsq, false, false);
  }

  // Diagonal M==N lives at element r = (lane&15) - 8*(lane>>4) on lanes 0-7, 24-31.
  {
    int r = m - kb;
    float dv = 0.f, nv = 0.f;
#pragma unroll
    for (int rr = 0; rr < 8; ++rr)
      if (rr == r) { dv = cdot[rr]; nv = cnsq[rr]; }
    if (r >= 0 && r < 8) { sdot[m] = dv; snsq[m] = nv; }
  }
  __syncthreads();

  if (lane < 16) {
    int node = n0 + lane;
    float nrm = fmaxf(sqrtf(snsq[lane]), 1e-8f);
    sw[lane] = sdot[lane] / (nrm * enorm[node]);
  }
  __syncthreads();

  // Scale x in place (input to next layer) and accumulate into acc (=d_out).
#pragma unroll 4
  for (int p = 0; p < 16; ++p) {
    float w = sw[p];
    size_t off = (size_t)(n0 + p) * DIM + lane * 2;
    float2 f = *(float2*)(x + off);
    f.x *= w; f.y *= w;
    *(float2*)(x + off) = f;
    float2 g = *(float2*)(acc + off);
    g.x += f.x; g.y += f.y;
    *(float2*)(acc + off) = g;
  }
}

// ---------------------------------------------------------------- launcher
extern "C" void kernel_launch(void* const* d_in, const int* in_sizes, int n_in,
                              void* d_out, int out_size, void* d_ws, size_t ws_size,
                              hipStream_t stream) {
  const float* ue   = (const float*)d_in[0];   // user_emb [NU, 64]
  const float* ie   = (const float*)d_in[1];   // item_emb [NI, 64]
  const int*   rows = (const int*)d_in[2];     // edge_rows [NE]
  const int*   cols = (const int*)d_in[3];     // edge_cols [NE]
  // d_in[4] = n_layers (scalar 3, hardcoded below)
  float* out = (float*)d_out;                  // acc, [NN, 64]
  float* ws  = (float*)d_ws;

  float* vals = ws;                            // NE
  float* du   = vals + NE;                     // NU
  float* di   = du + NU;                       // NI
  float* en   = di + NI;                       // NN
  float* xA   = en + NN;                       // NN*DIM
  float* xB   = xA + (size_t)NN * DIM;         // NN*DIM

  const int T = 256;
  const long ND = (long)NN * DIM;

  zero_kernel<<<(NU + T - 1) / T, T, 0, stream>>>(du, NU);
  zero_kernel<<<(NI + T - 1) / T, T, 0, stream>>>(di, NI);
  zero_kernel<<<(ND + T - 1) / T, T, 0, stream>>>(out, ND);

  degree_kernel<<<(NE + T - 1) / T, T, 0, stream>>>(rows, cols, du, di);
  vals_kernel<<<(NE + T - 1) / T, T, 0, stream>>>(rows, cols, du, di, vals);
  ego_norm_kernel<<<((long)NN * 32 + T - 1) / T, T, 0, stream>>>(ue, ie, en);

  float* xin  = nullptr;
  float* xout = xA;
  for (int l = 0; l < 3; ++l) {
    zero_kernel<<<(ND + T - 1) / T, T, 0, stream>>>(xout, ND);
    const float* su = (l == 0) ? ue : xin;
    const float* si = (l == 0) ? ie : (xin + (size_t)NU * DIM);
    long sp_threads = 2L * NE * 32;            // one wave per directed edge
    spmm_kernel<<<(sp_threads + T - 1) / T, T, 0, stream>>>(rows, cols, vals,
                                                            su, si, xout);
    reweight_kernel<<<NN / 16, 32, 0, stream>>>(xout, ue, ie, en, out);
    xin  = xout;
    xout = (xout == xA) ? xB : xA;
  }
}
